// Qwen3VLVisionAttention_57475252355432
// MI455X (gfx1250) — matrix operand order repaired and checked by reading, not compiled, not run on hardware
//
#include <hip/hip_runtime.h>
#include <hip/hip_bf16.h>

#define S_LEN   4096
#define NHEADS  16
#define HDIM    64
#define HIDDEN  1024

typedef __attribute__((ext_vector_type(16))) _Float16 v16h;
typedef __attribute__((ext_vector_type(8)))  float    v8f;
typedef __attribute__((ext_vector_type(4)))  _Float16 h4;
typedef __attribute__((ext_vector_type(8)))  _Float16 h8;
typedef __attribute__((__vector_size__(16))) int      i4v;

#define AS1 __attribute__((address_space(1)))
#define AS3 __attribute__((address_space(3)))

#if __has_builtin(__builtin_amdgcn_global_load_async_to_lds_b128)
#define HAVE_ASYNC_LDS 1
#else
#define HAVE_ASYNC_LDS 0
#endif

#if __has_builtin(__builtin_amdgcn_permlane16)
#define HAVE_PERMLANE16 1
#else
#define HAVE_PERMLANE16 0
#endif

union Frag { v16h h; unsigned u[8]; };

__device__ __forceinline__ v8f wmma_f16(v16h a, v16h b, v8f c) {
    return __builtin_amdgcn_wmma_f32_16x16x32_f16(
        false, a, false, b, (short)0, c, false, false);
}

// A-fragment (16x32 f16): lane holds row m = lane%16; VGPR v holds K pair.
__device__ __forceinline__ int a_frag_k(int lane, int v) {
    int kb = (lane < 16) ? 0 : 8;
    return (v < 4) ? (kb + 2 * v) : (16 + kb + 2 * (v - 4));
}
// B-fragment (32x16 f16): lane holds col n = lane%16; same K pairs as A-fragment.
__device__ __forceinline__ int b_frag_k(int lane, int v) {
    int kb = (lane < 16) ? 0 : 8;
    return (v < 4) ? (kb + 2 * v) : (16 + kb + 2 * (v - 4));
}

// 16-byte global -> LDS stage: async DMA path (ASYNCcnt) with vector fallback.
__device__ __forceinline__ void stage16(const _Float16* g, _Float16* l) {
#if HAVE_ASYNC_LDS
    __builtin_amdgcn_global_load_async_to_lds_b128(
        (AS1 i4v*)g, (AS3 i4v*)l, 0, 0);
#else
    *(h8*)l = *(const h8*)g;
#endif
}
__device__ __forceinline__ void wait_stage() {
#if HAVE_ASYNC_LDS
#if __has_builtin(__builtin_amdgcn_s_wait_asynccnt)
    __builtin_amdgcn_s_wait_asynccnt(0);
#else
    asm volatile("s_wait_asynccnt 0x0" ::: "memory");
#endif
#endif
}

// ---- width-16 butterfly reductions via v_permlane16_b32 (VALU, not LDS) ----
__device__ __forceinline__ float pl16x(float x, unsigned s0, unsigned s1) {
#if HAVE_PERMLANE16
    unsigned u = __builtin_bit_cast(unsigned, x);
    unsigned r = __builtin_amdgcn_permlane16(u, u, s0, s1, false, false);
    return __builtin_bit_cast(float, r);
#else
    (void)s0; (void)s1;
    return x;
#endif
}
__device__ __forceinline__ float red16_max(float x) {
#if HAVE_PERMLANE16
    x = fmaxf(x, pl16x(x, 0x67452301u, 0xEFCDAB89u));  // xor 1
    x = fmaxf(x, pl16x(x, 0x54761032u, 0xDCFE98BAu));  // xor 2
    x = fmaxf(x, pl16x(x, 0x32107654u, 0xBA98FEDCu));  // xor 4
    x = fmaxf(x, pl16x(x, 0xFEDCBA98u, 0x76543210u));  // xor 8
#else
    for (int off = 1; off < 16; off <<= 1) x = fmaxf(x, __shfl_xor(x, off, 16));
#endif
    return x;
}
__device__ __forceinline__ float red16_sum(float x) {
#if HAVE_PERMLANE16
    x += pl16x(x, 0x67452301u, 0xEFCDAB89u);
    x += pl16x(x, 0x54761032u, 0xDCFE98BAu);
    x += pl16x(x, 0x32107654u, 0xBA98FEDCu);
    x += pl16x(x, 0xFEDCBA98u, 0x76543210u);
#else
    for (int off = 1; off < 16; off <<= 1) x += __shfl_xor(x, off, 16);
#endif
    return x;
}

// ---------------------------------------------------------------------------
// One-time conversion kernels (HBM-bound, ~1 us total at 23.3 TB/s)
// ---------------------------------------------------------------------------
__global__ __launch_bounds__(256)
void cvt_f16_kernel(const float* __restrict__ in, _Float16* __restrict__ out, int n)
{
    int i = (blockIdx.x * 256 + threadIdx.x) * 4;
    if (i < n) {
        float4 v = *(const float4*)(in + i);
        h4 h = { (_Float16)v.x, (_Float16)v.y, (_Float16)v.z, (_Float16)v.w };
        *(h4*)(out + i) = h;
    }
}

// WT[n][k] = (f16) W[k][n]  -- K-major weights so GEMM B tiles are
// verbatim-copyable (async) and B fragments read as contiguous f16 pairs.
__global__ __launch_bounds__(256)
void cvt_transpose_kernel(const float* __restrict__ W, _Float16* __restrict__ WT,
                          int K, int N)
{
    __shared__ float tile[32][33];
    int kb = blockIdx.y * 32, nb = blockIdx.x * 32;
    int tx = threadIdx.x & 31, ty = threadIdx.x >> 5;
    for (int r = ty; r < 32; r += 8)
        tile[r][tx] = W[(size_t)(kb + r) * N + nb + tx];
    __syncthreads();
    for (int r = ty; r < 32; r += 8)
        WT[(size_t)(nb + r) * K + kb + tx] = (_Float16)tile[tx][r];
}

// ---------------------------------------------------------------------------
// GEMM: C[M,N] = A[M,K] (f16) * BT[N,K]^T (f16) + bias
// Block tile 128x128, 8 waves in 4(M) x 2(N); wave tile 32x64 = 2x4 WMMA tiles.
// Double-buffered LDS tiles; async DMA of tile i+1 overlaps WMMA on tile i.
// ---------------------------------------------------------------------------
template<bool OUT_HALF>
__global__ __launch_bounds__(256)
void gemm_f16_kernel(const _Float16* __restrict__ A,
                     const _Float16* __restrict__ BT,
                     const float* __restrict__ bias,
                     void* __restrict__ Cptr,
                     int M, int N, int K)
{
    __shared__ __align__(16) _Float16 ldsA[2][128 * 32];  // [buf][row][k]
    __shared__ __align__(16) _Float16 ldsB[2][128 * 32];  // [buf][col][k]

    const int nb = blockIdx.x * 128;
    const int mb = blockIdx.y * 128;
    const int tid  = threadIdx.x;
    const int wid  = tid >> 5;
    const int lane = tid & 31;
    const int waveM = wid & 3;     // 4 waves along M, 32 rows each
    const int waveN = wid >> 2;    // 2 waves along N, 64 cols each
    const int n_    = lane & 15;

    // staging pattern: 128 rows x 32 halfs = 512 x 16B chunks, 2 per thread
    const int srow0 = tid >> 2;
    const int scol  = (tid & 3) << 3;

    v8f acc[2][4];
    for (int mt = 0; mt < 2; ++mt)
        for (int nt = 0; nt < 4; ++nt)
            acc[mt][nt] = {};

    auto stage_tiles = [&](int kc, int buf) {
        for (int i = 0; i < 2; ++i) {
            int row = srow0 + i * 64;
            stage16(A  + (size_t)(mb + row) * K + kc + scol, &ldsA[buf][row * 32 + scol]);
            stage16(BT + (size_t)(nb + row) * K + kc + scol, &ldsB[buf][row * 32 + scol]);
        }
    };

    const int nchunks = K >> 5;
    stage_tiles(0, 0);

    for (int ic = 0; ic < nchunks; ++ic) {
        const int buf = ic & 1;
        wait_stage();
        __syncthreads();
        if (ic + 1 < nchunks)
            stage_tiles((ic + 1) << 5, buf ^ 1);   // DMA next tile during compute

        const _Float16* la = ldsA[buf];
        const _Float16* lb = ldsB[buf];
        Frag af[2];
        for (int mt = 0; mt < 2; ++mt) {
            int row = waveM * 32 + mt * 16 + n_;
            for (int v = 0; v < 8; ++v)
                af[mt].u[v] = *(const unsigned*)(&la[row * 32 + a_frag_k(lane, v)]);
        }
        Frag bf[4];
        for (int nt = 0; nt < 4; ++nt) {
            int col = waveN * 64 + nt * 16 + n_;
            for (int v = 0; v < 8; ++v)
                bf[nt].u[v] = *(const unsigned*)(&lb[col * 32 + b_frag_k(lane, v)]);
        }
        for (int mt = 0; mt < 2; ++mt)
            for (int nt = 0; nt < 4; ++nt)
                acc[mt][nt] = wmma_f16(af[mt].h, bf[nt].h, acc[mt][nt]);
    }

    for (int mt = 0; mt < 2; ++mt) {
        for (int nt = 0; nt < 4; ++nt) {
            int col = nb + waveN * 64 + nt * 16 + n_;
            float bv = bias[col];
            for (int v = 0; v < 8; ++v) {
                int row = mb + waveM * 32 + mt * 16 + v + ((lane < 16) ? 0 : 8);
                float val = acc[mt][nt][v] + bv;
                if (OUT_HALF)
                    ((_Float16*)Cptr)[(size_t)row * N + col] = (_Float16)val;
                else
                    ((float*)Cptr)[(size_t)row * N + col] = val;
            }
        }
    }
}

// ---------------------------------------------------------------------------
// RoPE + split qkv[S][3][H][D] (f16) into head-major q/k/v [H][S][D] f16
// ---------------------------------------------------------------------------
__global__ __launch_bounds__(256)
void rope_split_kernel(const _Float16* __restrict__ qkv,
                       const float* __restrict__ cosp,
                       const float* __restrict__ sinp,
                       _Float16* __restrict__ qh,
                       _Float16* __restrict__ kh,
                       _Float16* __restrict__ vh)
{
    int idx = blockIdx.x * 256 + threadIdx.x;   // [0, S*HIDDEN)
    int s  = idx >> 10;
    int hd = idx & 1023;
    int h  = hd >> 6;
    int d  = hd & 63;

    const _Float16* base = qkv + (size_t)s * (3 * HIDDEN);
    float q = (float)base[h * 64 + d];
    float k = (float)base[HIDDEN + h * 64 + d];
    float v = (float)base[2 * HIDDEN + h * 64 + d];
    float c  = cosp[s * 64 + d];
    float sn = sinp[s * 64 + d];
    int   dr  = (d < 32) ? d + 32 : d - 32;
    float sgn = (d < 32) ? -1.0f : 1.0f;
    float qr = sgn * (float)base[h * 64 + dr];
    float kr = sgn * (float)base[HIDDEN + h * 64 + dr];

    size_t o = ((size_t)h * S_LEN + s) * 64 + d;
    qh[o] = (_Float16)(q * c + qr * sn);
    kh[o] = (_Float16)(k * c + kr * sn);
    vh[o] = (_Float16)v;
}

// ---------------------------------------------------------------------------
// Flash attention: block = (head, 128 query rows), 8 waves x 16 rows each.
// Online softmax over 32-key blocks; QK^T and PV via v_wmma_f32_16x16x32_f16.
// Double-buffered K/V tiles: async DMA of block i+1 overlaps WMMA on block i.
// Softmax row reductions via v_permlane16_b32 (VALU pipe, not LDS).
// ---------------------------------------------------------------------------
__global__ __launch_bounds__(256)
void attn_kernel(const _Float16* __restrict__ qh,
                 const _Float16* __restrict__ kh,
                 const _Float16* __restrict__ vh,
                 _Float16* __restrict__ oh)
{
    __shared__ __align__(16) _Float16 ldsK[2][32 * 64];   // [buf][key][dim]
    __shared__ __align__(16) _Float16 ldsVT[2][64 * 32];  // [buf][dim][key]
    __shared__ __align__(16) _Float16 ldsP[8][16 * 32];   // per-wave P scratch

    const int h  = blockIdx.x >> 5;            // 32 query blocks per head
    const int qb = (blockIdx.x & 31) * 128;
    const int tid  = threadIdx.x;
    const int wid  = tid >> 5;
    const int lane = tid & 31;
    const int n_   = lane & 15;
    const int qrow0 = qb + wid * 16;

    // Q fragments: 16 rows x 64 head-dim = two K=32 chunks
    Frag qf[2];
    const _Float16* qbase = qh + ((size_t)h * S_LEN + qrow0) * 64;
    for (int c = 0; c < 2; ++c)
        for (int v = 0; v < 8; ++v) {
            int k = c * 32 + a_frag_k(lane, v);
            qf[c].u[v] = *(const unsigned*)(qbase + (size_t)n_ * 64 + k);
        }

    v8f o[4];
    for (int t = 0; t < 4; ++t) o[t] = {};
    float m_r[8], l_r[8];
    for (int v = 0; v < 8; ++v) { m_r[v] = -1e30f; l_r[v] = 0.0f; }

    const _Float16* kbase = kh + (size_t)h * S_LEN * 64;
    const _Float16* vbase = vh + (size_t)h * S_LEN * 64;
    const float scale = 0.125f;   // 1/sqrt(64)

    // cooperative-load indices: 2048 halfs per 32x64 tile, 8 halfs per thread
    const int cr = tid >> 3;           // key row 0..31
    const int cc = (tid & 7) << 3;     // dim col 0,8,..,56

    auto stage_kv = [&](int kb, int buf) {
        // K tile: async DMA straight into LDS (row-major, verbatim)
        stage16(kbase + (size_t)(kb + cr) * 64 + cc, &ldsK[buf][cr * 64 + cc]);
        // V tile: transposed in flight (dim-major for PV B-fragments)
        h8 vv = *(const h8*)(vbase + (size_t)(kb + cr) * 64 + cc);
        for (int j = 0; j < 8; ++j)
            ldsVT[buf][(cc + j) * 32 + cr] = vv[j];
    };

    stage_kv(0, 0);

    for (int kb = 0; kb < S_LEN; kb += 32) {
        const int buf = (kb >> 5) & 1;
        wait_stage();
        __syncthreads();
        if (kb + 32 < S_LEN)
            stage_kv(kb + 32, buf ^ 1);           // stage next block during compute

        const _Float16* lk = ldsK[buf];
        const _Float16* lv = ldsVT[buf];

        // ---- S = Q K^T  (two 16-key tiles, reduce over 64-dim in 2 chunks) ----
        v8f s0 = {}, s1 = {};
        for (int c = 0; c < 2; ++c) {
            Frag b0, b1;
            for (int v = 0; v < 8; ++v) {
                int kk = c * 32 + b_frag_k(lane, v);
                b0.u[v] = *(const unsigned*)(&lk[n_ * 64 + kk]);
                b1.u[v] = *(const unsigned*)(&lk[(16 + n_) * 64 + kk]);
            }
            s0 = wmma_f16(qf[c].h, b0.h, s0);
            s1 = wmma_f16(qf[c].h, b1.h, s1);
        }

        // ---- online softmax (row = VGPR index, cols striped across 16 lanes) ----
        float p0[8], p1[8], f[8];
        for (int v = 0; v < 8; ++v) {
            float a0 = s0[v] * scale, a1 = s1[v] * scale;
            float m  = red16_max(fmaxf(a0, a1));
            float mn = fmaxf(m_r[v], m);
            float fv = __expf(m_r[v] - mn);
            float e0 = __expf(a0 - mn);
            float e1 = __expf(a1 - mn);
            float rs = red16_sum(e0 + e1);
            l_r[v] = l_r[v] * fv + rs;
            m_r[v] = mn;
            f[v] = fv; p0[v] = e0; p1[v] = e1;
        }
        for (int t = 0; t < 4; ++t)
            for (int v = 0; v < 8; ++v)
                o[t][v] *= f[v];

        // ---- transpose P through per-wave LDS into A-fragment layout ----
        _Float16* myP = ldsP[wid];
        for (int v = 0; v < 8; ++v) {
            int row = v + ((lane < 16) ? 0 : 8);
            myP[row * 32 + n_]      = (_Float16)p0[v];
            myP[row * 32 + 16 + n_] = (_Float16)p1[v];
        }
        Frag pf;
        for (int v = 0; v < 8; ++v)
            pf.u[v] = *(const unsigned*)(&myP[n_ * 32 + a_frag_k(lane, v)]);

        // ---- O += P V  (4 dim tiles of 16) ----
        for (int t = 0; t < 4; ++t) {
            Frag vf;
            for (int v = 0; v < 8; ++v)
                vf.u[v] = *(const unsigned*)(&lv[(t * 16 + n_) * 32 + b_frag_k(lane, v)]);
            o[t] = wmma_f16(pf.h, vf.h, o[t]);
        }
    }

    // ---- normalize and store to oh[S][HIDDEN] (f16, layout for proj GEMM) ----
    for (int t = 0; t < 4; ++t)
        for (int v = 0; v < 8; ++v) {
            int row = qrow0 + v + ((lane < 16) ? 0 : 8);
            int col = h * 64 + t * 16 + n_;
            oh[(size_t)row * HIDDEN + col] = (_Float16)(o[t][v] / l_r[v]);
        }
}

// ---------------------------------------------------------------------------
extern "C" void kernel_launch(void* const* d_in, const int* in_sizes, int n_in,
                              void* d_out, int out_size, void* d_ws, size_t ws_size,
                              hipStream_t stream)
{
    const float* hidden = (const float*)d_in[0];
    // d_in[1] = cu_seqlens: dead in the reference (full softmax), ignored.
    const float* cosp  = (const float*)d_in[2];
    const float* sinp  = (const float*)d_in[3];
    const float* Wqkv  = (const float*)d_in[4];
    const float* bqkv  = (const float*)d_in[5];
    const float* Wproj = (const float*)d_in[6];
    const float* bproj = (const float*)d_in[7];
    float* out = (float*)d_out;

    _Float16* ws = (_Float16*)d_ws;
    _Float16* hidden_h = ws;                                   // 4M halfs
    _Float16* WqkvT    = hidden_h + (size_t)S_LEN * HIDDEN;    // 3M halfs [3072][1024]
    _Float16* WprojT   = WqkvT + (size_t)3 * HIDDEN * HIDDEN;  // 1M halfs [1024][1024]
    _Float16* qkv      = WprojT + (size_t)HIDDEN * HIDDEN;     // 12M halfs
    _Float16* qh       = qkv + (size_t)S_LEN * 3 * HIDDEN;     // 4M halfs each
    _Float16* kh       = qh + (size_t)NHEADS * S_LEN * HDIM;
    _Float16* vh       = kh + (size_t)NHEADS * S_LEN * HDIM;
    _Float16* oh       = vh + (size_t)NHEADS * S_LEN * HDIM;

    // 0) one-time f32->f16 conversions (weights K-major transposed)
    cvt_f16_kernel<<<(S_LEN * HIDDEN) / 1024, 256, 0, stream>>>(
        hidden, hidden_h, S_LEN * HIDDEN);
    cvt_transpose_kernel<<<dim3(3 * HIDDEN / 32, HIDDEN / 32), 256, 0, stream>>>(
        Wqkv, WqkvT, HIDDEN, 3 * HIDDEN);
    cvt_transpose_kernel<<<dim3(HIDDEN / 32, HIDDEN / 32), 256, 0, stream>>>(
        Wproj, WprojT, HIDDEN, HIDDEN);

    // 1) QKV = hidden @ W_qkv + b_qkv   (f16 out)
    dim3 g1(3 * HIDDEN / 128, S_LEN / 128);
    gemm_f16_kernel<true><<<g1, 256, 0, stream>>>(
        hidden_h, WqkvT, bqkv, qkv, S_LEN, 3 * HIDDEN, HIDDEN);

    // 2) RoPE + head-major split
    rope_split_kernel<<<(S_LEN * HIDDEN) / 256, 256, 0, stream>>>(
        qkv, cosp, sinp, qh, kh, vh);

    // 3) Flash attention per (head, 128-row query block)
    attn_kernel<<<NHEADS * (S_LEN / 128), 256, 0, stream>>>(qh, kh, vh, oh);

    // 4) out = oh @ W_proj + b_proj   (f32 out)
    dim3 g2(HIDDEN / 128, S_LEN / 128);
    gemm_f16_kernel<false><<<g2, 256, 0, stream>>>(
        oh, WprojT, bproj, out, S_LEN, HIDDEN, HIDDEN);
}
